// DynamicGCN_54185307406456
// MI455X (gfx1250) — matrix operand order repaired and checked by reading, not compiled, not run on hardware
//
#include <hip/hip_runtime.h>

typedef __attribute__((ext_vector_type(16))) _Float16 v16h;
typedef __attribute__((ext_vector_type(8)))  _Float16 v8h;
typedef __attribute__((ext_vector_type(4)))  _Float16 v4h;
typedef __attribute__((ext_vector_type(8)))  float    v8f;
typedef __attribute__((ext_vector_type(4)))  float    v4f;

#define BB_  16
#define NN_  512
#define TT_  12
#define DD_  32
#define HH_  64
// pre-scale X rows by 32^-0.25 so S = Xs*Xs^T is already scaled by 1/sqrt(32)
#define XSCL 0.42044820762685725f

__device__ __forceinline__ v8f wmma_f16(v16h a, v16h b, v8f c) {
  return __builtin_amdgcn_wmma_f32_16x16x32_f16(false, a, false, b, (short)0, c,
                                                false, false);
}

// XOR-16 lane exchange within wave32 (group-of-32 swizzle, xor_mask=0x10)
__device__ __forceinline__ float swz16(float x) {
  return __int_as_float(__builtin_amdgcn_ds_swizzle(__float_as_int(x), 0x401f));
}

// A-operand (16x32 f16): lane L holds row r0+(L&15); V0-3 = K (L>>4)*8+{0..7},
// V4-7 = K 16+(L>>4)*8+{0..7}. Two contiguous 16B LDS loads per lane.
__device__ __forceinline__ v16h ldA(const _Float16* base, int stride, int r0,
                                    int kofs, int lane) {
  const _Float16* p = base + (r0 + (lane & 15)) * stride + kofs + ((lane >> 4) << 3);
  v8h lo = *(const v8h*)p;
  v8h hi = *(const v8h*)(p + 16);
  v16h a;
#pragma unroll
  for (int i = 0; i < 8; ++i) { a[i] = lo[i]; a[i + 8] = hi[i]; }
  return a;
}

// B-operand (32x16 f16): lane L holds column c0+(L&15); V0-3 = K (L>>4)*8+{0..7},
// V4-7 = K 16+(L>>4)*8+{0..7}. Two contiguous 16B LDS loads per lane.
__device__ __forceinline__ v16h ldB(const _Float16* base, int stride, int c0,
                                    int kofs, int lane) {
  const _Float16* p = base + (c0 + (lane & 15)) * stride + kofs + ((lane >> 4) << 3);
  v8h lo = *(const v8h*)p;
  v8h hi = *(const v8h*)(p + 16);
  v16h b;
#pragma unroll
  for (int i = 0; i < 8; ++i) { b[i] = lo[i]; b[i + 8] = hi[i]; }
  return b;
}

__global__ void __launch_bounds__(256)
dyn_gcn_kernel(const float* __restrict__ x, const float* __restrict__ W1,
               const float* __restrict__ b1, const float* __restrict__ W2,
               const float* __restrict__ b2, float* __restrict__ out) {
  __shared__ alignas(32) _Float16 Xs[NN_ * DD_];        // 32 KB, scaled, row-major
  __shared__ alignas(32) _Float16 Xt[DD_ * NN_];        // 32 KB, unscaled, transposed
  __shared__ alignas(32) _Float16 Ht[HH_ * NN_];        // 64 KB, h transposed
  __shared__ alignas(32) _Float16 W1t[HH_ * DD_];       // 4 KB
  __shared__ alignas(32) _Float16 W2t[HH_ * HH_];       // 8 KB
  __shared__ alignas(32) float    b1s[HH_];
  __shared__ alignas(32) float    b2s[HH_];
  __shared__ alignas(32) float    stats[8 * 16];        // per-wave 1/rowsum
  __shared__ alignas(32) _Float16 stage[8 * 16 * HH_];  // 16 KB, per-wave C->A bounce

  const int tid  = threadIdx.x;
  const int lane = tid & 31;
  const int wv   = tid >> 5;
  const int ln   = lane & 15;
  const int hi   = lane >> 4;
  const int bb   = blockIdx.x / TT_;
  const int tf   = blockIdx.x % TT_;

  // ---------- stage inputs into LDS ----------
  for (int i = tid; i < NN_ * 8; i += 256) {        // 8 float4 per row of X
    int n = i >> 3, q = i & 7;
    const float* src = x + (((size_t)(bb * NN_ + n) * TT_ + tf) << 5) + (q << 2);
    v4f xv = *(const v4f*)src;
    v4h xs;
#pragma unroll
    for (int k = 0; k < 4; ++k) xs[k] = (_Float16)(xv[k] * XSCL);
    *(v4h*)(Xs + n * DD_ + (q << 2)) = xs;
#pragma unroll
    for (int k = 0; k < 4; ++k) Xt[((q << 2) + k) * NN_ + n] = (_Float16)xv[k];
  }
  for (int i = tid; i < DD_ * HH_; i += 256) {      // W1t[h][d] = W1[d][h]
    int d = i >> 6, h = i & 63;
    W1t[h * DD_ + d] = (_Float16)W1[i];
  }
  for (int i = tid; i < HH_ * HH_; i += 256) {      // W2t[k][h] = W2[h][k]
    int r = i >> 6, k = i & 63;
    W2t[k * HH_ + r] = (_Float16)W2[i];
  }
  if (tid < HH_) { b1s[tid] = b1[tid]; b2s[tid] = b2[tid]; }
  __syncthreads();

  // =================== PASS 1: h = relu(((P@X)/sum)@W1 + b1) ===================
  for (int mb = wv; mb < 32; mb += 8) {
    const int m0 = mb * 16;
    const v16h bm = ldB(Xs, DD_, m0, 0, lane);      // X m-block as B (X^T)

    // --- row max over 512 scaled scores (lane L owns row L&15) ---
    float mrow = -1e30f;
    for (int j = 0; j < 32; ++j) {
      v16h a = ldA(Xs, DD_, j * 16, 0, lane);
      v8f c = {};
      c = wmma_f16(a, bm, c);                       // T_j = X_j @ X_m^T (scaled)
#pragma unroll
      for (int v = 0; v < 8; ++v) mrow = fmaxf(mrow, c[v]);
    }
    mrow = fmaxf(mrow, swz16(mrow));

    // --- O' = P @ X  (unnormalized), K streamed 32 at a time ---
    v8f o0 = {}, o1 = {};
    float sum = 0.f;
    for (int u = 0; u < 16; ++u) {
      v16h a0 = ldA(Xs, DD_, (2 * u) * 16, 0, lane);
      v16h a1 = ldA(Xs, DD_, (2 * u + 1) * 16, 0, lane);
      v8f c0 = {}, c1 = {};
      c0 = wmma_f16(a0, bm, c0);
      c1 = wmma_f16(a1, bm, c1);
      v16h p;                                        // T-layout == A-layout halves
#pragma unroll
      for (int v = 0; v < 8; ++v) {
        float e0 = __expf(c0[v] - mrow);
        float e1 = __expf(c1[v] - mrow);
        sum += e0 + e1;
        p[v] = (_Float16)e0;
        p[v + 8] = (_Float16)e1;
      }
      o0 = wmma_f16(p, ldB(Xt, NN_, 0, u * 32, lane), o0);
      o1 = wmma_f16(p, ldB(Xt, NN_, 16, u * 32, lane), o1);
    }
    sum += swz16(sum);
    if (lane < 16) stats[wv * 16 + lane] = 1.0f / sum;          // in-order LDS
    v8f rs = *(const v8f*)(stats + wv * 16 + hi * 8);

    // --- stage (O'/sum) as f16 row-major, re-read as A, multiply W1t ---
    _Float16* stg = stage + wv * 16 * HH_;
#pragma unroll
    for (int v = 0; v < 8; ++v) {
      int r = v + 8 * hi;
      stg[r * HH_ + ln]      = (_Float16)(o0[v] * rs[v]);
      stg[r * HH_ + 16 + ln] = (_Float16)(o1[v] * rs[v]);
    }
    v16h pa = ldA(stg, HH_, 0, 0, lane);
#pragma unroll
    for (int ct = 0; ct < 4; ++ct) {
      v8f hc = {};
      hc = wmma_f16(pa, ldB(W1t, DD_, ct * 16, 0, lane), hc);
      float bv = b1s[ct * 16 + ln];
      v8h hv;
#pragma unroll
      for (int v = 0; v < 8; ++v) hv[v] = (_Float16)fmaxf(hc[v] + bv, 0.f);
      *(v8h*)(Ht + (ct * 16 + ln) * NN_ + m0 + 8 * hi) = hv;    // h transposed
    }
  }
  __syncthreads();

  // =================== PASS 2: out = ((P@h)/sum)@W2 + b2 ===================
  for (int mb = wv; mb < 32; mb += 8) {
    const int m0 = mb * 16;
    const v16h bm = ldB(Xs, DD_, m0, 0, lane);

    float mrow = -1e30f;
    for (int j = 0; j < 32; ++j) {
      v16h a = ldA(Xs, DD_, j * 16, 0, lane);
      v8f c = {};
      c = wmma_f16(a, bm, c);
#pragma unroll
      for (int v = 0; v < 8; ++v) mrow = fmaxf(mrow, c[v]);
    }
    mrow = fmaxf(mrow, swz16(mrow));

    v8f o[4] = {{}, {}, {}, {}};
    float sum = 0.f;
    for (int u = 0; u < 16; ++u) {
      v16h a0 = ldA(Xs, DD_, (2 * u) * 16, 0, lane);
      v16h a1 = ldA(Xs, DD_, (2 * u + 1) * 16, 0, lane);
      v8f c0 = {}, c1 = {};
      c0 = wmma_f16(a0, bm, c0);
      c1 = wmma_f16(a1, bm, c1);
      v16h p;
#pragma unroll
      for (int v = 0; v < 8; ++v) {
        float e0 = __expf(c0[v] - mrow);
        float e1 = __expf(c1[v] - mrow);
        sum += e0 + e1;
        p[v] = (_Float16)e0;
        p[v + 8] = (_Float16)e1;
      }
#pragma unroll
      for (int ct = 0; ct < 4; ++ct)
        o[ct] = wmma_f16(p, ldB(Ht, NN_, ct * 16, u * 32, lane), o[ct]);
    }
    sum += swz16(sum);
    if (lane < 16) stats[wv * 16 + lane] = 1.0f / sum;
    v8f rs = *(const v8f*)(stats + wv * 16 + hi * 8);

    _Float16* stg = stage + wv * 16 * HH_;
#pragma unroll
    for (int ct = 0; ct < 4; ++ct)
#pragma unroll
      for (int v = 0; v < 8; ++v)
        stg[(v + 8 * hi) * HH_ + ct * 16 + ln] = (_Float16)(o[ct][v] * rs[v]);

    v8f oc[4] = {{}, {}, {}, {}};
#pragma unroll
    for (int u2 = 0; u2 < 2; ++u2) {
      v16h pa = ldA(stg, HH_, 0, u2 * 32, lane);
#pragma unroll
      for (int ct = 0; ct < 4; ++ct)
        oc[ct] = wmma_f16(pa, ldB(W2t, HH_, ct * 16, u2 * 32, lane), oc[ct]);
    }
#pragma unroll
    for (int ct = 0; ct < 4; ++ct) {
      float bv = b2s[ct * 16 + ln];
#pragma unroll
      for (int v = 0; v < 8; ++v) {
        int n = m0 + v + 8 * hi;
        out[((size_t)(bb * NN_ + n) * TT_ + tf) * HH_ + ct * 16 + ln] = oc[ct][v] + bv;
      }
    }
  }
}

extern "C" void kernel_launch(void* const* d_in, const int* in_sizes, int n_in,
                              void* d_out, int out_size, void* d_ws, size_t ws_size,
                              hipStream_t stream) {
  (void)in_sizes; (void)n_in; (void)out_size; (void)d_ws; (void)ws_size;
  const float* x  = (const float*)d_in[0];
  const float* W1 = (const float*)d_in[1];
  const float* b1 = (const float*)d_in[2];
  const float* W2 = (const float*)d_in[3];
  const float* b2 = (const float*)d_in[4];
  float* out = (float*)d_out;
  hipLaunchKernelGGL(dyn_gcn_kernel, dim3(BB_ * TT_), dim3(256), 0, stream,
                     x, W1, b1, W2, b2, out);
}
